// TextEmbedding_45681272160517
// MI455X (gfx1250) — matrix operand order repaired and checked by reading, not compiled, not run on hardware
//
#include <hip/hip_runtime.h>
#include <hip/hip_bf16.h>
#include <math.h>
#include <stdint.h>

// ---- problem constants (match reference) ----
#define BB    8
#define SEQ   4096
#define DIM   512
#define INTER 1024
#define LAYERS 4
#define TTEXT 512

typedef __attribute__((ext_vector_type(16))) __bf16 v16bf;
typedef __attribute__((ext_vector_type(8)))  float  v8f;

// ============================================================
// Pack a row-major f32 weight W[K,N] into wave32 WMMA B-fragment
// layout (bf16): P[((ntile*(K/32)+kt)*32 + lane)*16 + e]
//   lane 0..15  -> column n = ntile*16 + lane,  K = kt*32 + e
//   lane 16..31 -> column n = ntile*16 + lane-16, K = kt*32 + 16 + e
// ============================================================
__global__ void pack_b_kernel(const float* __restrict__ W, __bf16* __restrict__ P,
                              int K, int N) {
    int i = blockIdx.x * 256 + threadIdx.x;
    if (i >= K * N) return;
    int e    = i & 15;
    int lane = (i >> 4) & 31;
    int kt   = (i >> 9) % (K >> 5);
    int nt   = i / ((K >> 5) << 9);
    int n = nt * 16 + (lane & 15);
    int k = kt * 32 + ((lane >> 4) << 4) + e;
    P[i] = (__bf16)W[(size_t)k * N + n];
}

// ============================================================
// Embedding + absolute positional (cos|sin) encoding + mask
// ============================================================
__global__ void embed_kernel(const int* __restrict__ text, const int* __restrict__ seq_len,
                             const float* __restrict__ emb, float* __restrict__ X) {
    size_t i = (size_t)blockIdx.x * 256 + threadIdx.x;     // B*S*D elements
    int d = (int)(i & (DIM - 1));
    int s = (int)((i >> 9) & (SEQ - 1));
    int b = (int)(i >> 21);
    int sl = seq_len[b];
    float val = 0.0f;
    if (s < TTEXT && s < sl) {
        int id = text[b * TTEXT + s] + 1;
        int di = (d < 256) ? d : (d - 256);
        float f0 = expf(-9.210340371976184f * (float)di * (1.0f / 256.0f)); // theta^-(d/256)
        float ang = (float)s * f0;
        float fr = (d < 256) ? cosf(ang) : sinf(ang);
        val = emb[(size_t)id * DIM + d] + fr;
    }
    X[i] = val;
}

// ============================================================
// Depthwise conv (k=7, pad 3) + bias + LayerNorm over channels
// ============================================================
__global__ void dwln_kernel(const float* __restrict__ X, float* __restrict__ Y,
                            const float* __restrict__ dw_w, const float* __restrict__ dw_b,
                            const float* __restrict__ ln_g, const float* __restrict__ ln_b) {
    int bs = blockIdx.x;
    int b = bs >> 12;
    int s = bs & (SEQ - 1);
    int tid = threadIdx.x;

    float v[2];
#pragma unroll
    for (int i = 0; i < 2; ++i) {
        int c = tid + i * 256;
        float acc = dw_b[c];
#pragma unroll
        for (int t = 0; t < 7; ++t) {
            int ss = s + t - 3;
            if (ss >= 0 && ss < SEQ)
                acc += X[((size_t)b * SEQ + ss) * DIM + c] * dw_w[t * DIM + c];
        }
        v[i] = acc;
    }
    __shared__ float s1[256], s2[256];
    s1[tid] = v[0] + v[1];
    s2[tid] = v[0] * v[0] + v[1] * v[1];
    __syncthreads();
    for (int off = 128; off > 0; off >>= 1) {
        if (tid < off) { s1[tid] += s1[tid + off]; s2[tid] += s2[tid + off]; }
        __syncthreads();
    }
    float mean = s1[0] * (1.0f / DIM);
    float var  = s2[0] * (1.0f / DIM) - mean * mean;
    float rstd = rsqrtf(var + 1e-6f);
#pragma unroll
    for (int i = 0; i < 2; ++i) {
        int c = tid + i * 256;
        Y[(size_t)bs * DIM + c] = (v[i] - mean) * rstd * ln_g[c] + ln_b[c];
    }
}

// ------------------------------------------------------------
// Issue one block-wide async copy: 4 B-fragments (4 KB) of K-step
// kt into LDS buffer at lds_base.  One b128 per thread (256 thr).
// Tracked by ASYNCcnt.
// ------------------------------------------------------------
__device__ __forceinline__ void issue_bfrag_async(const __bf16* __restrict__ Bp, int nt0,
                                                  int KT32, int kt, uint32_t lds_base,
                                                  int tid) {
    int j = tid >> 6;          // fragment 0..3
    int o = tid & 63;          // 16B chunk within fragment
    const __bf16* g = Bp + (((size_t)(nt0 + j) * KT32 + kt) << 9) + (size_t)o * 8;
    uint32_t l = lds_base + (uint32_t)tid * 16;
    asm volatile("global_load_async_to_lds_b128 %0, %1, off"
                 :: "v"(l), "v"(g) : "memory");
}

// ============================================================
// GEMM1: H[M,1024] = GELU(Y[M,512] @ W1 + b1)   bf16 WMMA, f32 acc
// block tile 256Mx64N (8 waves x 32Mx64N); B staged via async->LDS
// ============================================================
__global__ void gemm1_gelu_kernel(const float* __restrict__ A, const __bf16* __restrict__ Bp,
                                  const float* __restrict__ bias, float* __restrict__ H) {
    const int K = DIM, N = INTER, KT = K / 32, NBLK = N / 64;
    __shared__ __align__(16) char sB[2][4096];

    int tid  = threadIdx.x;
    int lane = tid & 31;
    int w    = tid >> 5;
    int blk_m = blockIdx.x / NBLK;
    int blk_n = blockIdx.x % NBLK;
    int m0 = blk_m * 256 + w * 32;
    int n0 = blk_n * 64;
    int nt0 = n0 >> 4;
    int col  = lane & 15;
    int half = lane >> 4;

    v8f acc0[4], acc1[4];
#pragma unroll
    for (int j = 0; j < 4; ++j) {
        float bv = bias[n0 + j * 16 + col];
#pragma unroll
        for (int v = 0; v < 8; ++v) { acc0[j][v] = bv; acc1[j][v] = bv; }
    }

    const float* arow0 = A + (size_t)(m0 + col) * K + half * 16;
    const float* arow1 = arow0 + (size_t)16 * K;
    uint32_t lds0 = (uint32_t)(uintptr_t)&sB[0][0];
    uint32_t lds1 = (uint32_t)(uintptr_t)&sB[1][0];

    issue_bfrag_async(Bp, nt0, KT, 0, lds0, tid);

    for (int kt = 0; kt < KT; ++kt) {
        asm volatile("s_wait_asynccnt 0x0" ::: "memory");
        __syncthreads();                                   // kt's B is in LDS

        const float* ap0 = arow0 + kt * 32;
        const float* ap1 = arow1 + kt * 32;
        v16bf a0, a1;
#pragma unroll
        for (int e = 0; e < 8; ++e) {
            a0[e] = (__bf16)ap0[e]; a0[e + 8] = (__bf16)ap0[e + 8];
            a1[e] = (__bf16)ap1[e]; a1[e + 8] = (__bf16)ap1[e + 8];
        }
        const char* buf = sB[kt & 1];
#pragma unroll
        for (int j = 0; j < 4; ++j) {
            v16bf bf = *(const v16bf*)(buf + j * 1024 + lane * 32);
            acc0[j] = __builtin_amdgcn_wmma_f32_16x16x32_bf16(
                false, a0, false, bf, (short)0, acc0[j], false, false);
            acc1[j] = __builtin_amdgcn_wmma_f32_16x16x32_bf16(
                false, a1, false, bf, (short)0, acc1[j], false, false);
        }
        __syncthreads();                                   // B regs consumed; buffer reusable
        if (kt + 1 < KT)
            issue_bfrag_async(Bp, nt0, KT, kt + 1, (kt & 1) ? lds0 : lds1, tid);
    }

#pragma unroll
    for (int j = 0; j < 4; ++j) {
        int n = n0 + j * 16 + col;
#pragma unroll
        for (int v = 0; v < 8; ++v) {
            int m = m0 + half * 8 + v;
            float x0 = acc0[j][v];
            float x1 = acc1[j][v];
            x0 = 0.5f * x0 * (1.0f + erff(x0 * 0.70710678118f));   // exact GELU
            x1 = 0.5f * x1 * (1.0f + erff(x1 * 0.70710678118f));
            H[(size_t)m * N + n] = x0;
            H[(size_t)(m + 16) * N + n] = x1;
        }
    }
}

// ============================================================
// GRN part 1: gx[b,c] = sqrt( sum_s H[b,s,c]^2 )
// ============================================================
__global__ void grn_sumsq_kernel(const float* __restrict__ H, float* __restrict__ gx) {
    int blk = blockIdx.x;
    int b  = blk >> 4;
    int c0 = (blk & 15) * 64;
    int cl = threadIdx.x & 63;
    int sl = threadIdx.x >> 6;
    int c = c0 + cl;
    const float* base = H + (size_t)b * SEQ * INTER + c;
    float sum = 0.0f;
    int s0 = sl * (SEQ / 4);
    for (int s = s0; s < s0 + SEQ / 4; ++s) {
        float v = base[(size_t)s * INTER];
        sum += v * v;
    }
    __shared__ float sm[256];
    sm[threadIdx.x] = sum;
    __syncthreads();
    if (sl == 0) {
        float t = sm[cl] + sm[cl + 64] + sm[cl + 128] + sm[cl + 192];
        gx[b * INTER + c] = sqrtf(t);
    }
}

// GRN part 2: alpha[b,c] = grn_g[c] * gx/(mean_c(gx)+1e-6) + 1
__global__ void grn_alpha_kernel(const float* __restrict__ gx, const float* __restrict__ grn_g,
                                 float* __restrict__ alpha) {
    int b = blockIdx.x;
    int tid = threadIdx.x;
    float gv[4]; float loc = 0.0f;
#pragma unroll
    for (int i = 0; i < 4; ++i) { gv[i] = gx[b * INTER + tid + i * 256]; loc += gv[i]; }
    __shared__ float sm[256];
    sm[tid] = loc;
    __syncthreads();
    for (int off = 128; off > 0; off >>= 1) {
        if (tid < off) sm[tid] += sm[tid + off];
        __syncthreads();
    }
    float inv = 1.0f / (sm[0] * (1.0f / INTER) + 1e-6f);
#pragma unroll
    for (int i = 0; i < 4; ++i) {
        int c = tid + i * 256;
        alpha[b * INTER + c] = grn_g[c] * (gv[i] * inv) + 1.0f;
    }
}

// ============================================================
// GEMM2 (GRN fused into A conversion):
//   X = mask( X + (H*alpha + grn_b)[M,1024] @ W2 + b2 )  in place
// ============================================================
__global__ void gemm2_res_kernel(const float* __restrict__ A, const __bf16* __restrict__ Bp,
                                 const float* __restrict__ bias, float* __restrict__ X,
                                 const int* __restrict__ seq_len,
                                 const float* __restrict__ alpha,
                                 const float* __restrict__ grn_b) {
    const int K = INTER, N = DIM, KT = K / 32, NBLK = N / 64;
    __shared__ __align__(16) char sB[2][4096];

    int tid  = threadIdx.x;
    int lane = tid & 31;
    int w    = tid >> 5;
    int blk_m = blockIdx.x / NBLK;
    int blk_n = blockIdx.x % NBLK;
    int m0 = blk_m * 256 + w * 32;
    int n0 = blk_n * 64;
    int nt0 = n0 >> 4;
    int col  = lane & 15;
    int half = lane >> 4;

    v8f acc0[4], acc1[4];
#pragma unroll
    for (int j = 0; j < 4; ++j) {
        float bv = bias[n0 + j * 16 + col];
#pragma unroll
        for (int v = 0; v < 8; ++v) { acc0[j][v] = bv; acc1[j][v] = bv; }
    }

    const float* arow0 = A + (size_t)(m0 + col) * K + half * 16;
    const float* arow1 = arow0 + (size_t)16 * K;
    // all 32 rows of this wave live in the same batch b = m0 >> 12
    const float* alr = alpha + ((size_t)(m0 >> 12) << 10) + half * 16;
    const float* gbr = grn_b + half * 16;
    uint32_t lds0 = (uint32_t)(uintptr_t)&sB[0][0];
    uint32_t lds1 = (uint32_t)(uintptr_t)&sB[1][0];

    issue_bfrag_async(Bp, nt0, KT, 0, lds0, tid);

    for (int kt = 0; kt < KT; ++kt) {
        asm volatile("s_wait_asynccnt 0x0" ::: "memory");
        __syncthreads();

        const float* ap0 = arow0 + kt * 32;
        const float* ap1 = arow1 + kt * 32;
        const float* alp = alr + kt * 32;
        const float* gbp = gbr + kt * 32;
        v16bf a0, a1;
#pragma unroll
        for (int e = 0; e < 8; ++e) {
            float sA = alp[e], sB2 = gbp[e];
            float sA2 = alp[e + 8], sB3 = gbp[e + 8];
            a0[e]     = (__bf16)(ap0[e] * sA + sB2);
            a0[e + 8] = (__bf16)(ap0[e + 8] * sA2 + sB3);
            a1[e]     = (__bf16)(ap1[e] * sA + sB2);
            a1[e + 8] = (__bf16)(ap1[e + 8] * sA2 + sB3);
        }
        const char* buf = sB[kt & 1];
#pragma unroll
        for (int j = 0; j < 4; ++j) {
            v16bf bf = *(const v16bf*)(buf + j * 1024 + lane * 32);
            acc0[j] = __builtin_amdgcn_wmma_f32_16x16x32_bf16(
                false, a0, false, bf, (short)0, acc0[j], false, false);
            acc1[j] = __builtin_amdgcn_wmma_f32_16x16x32_bf16(
                false, a1, false, bf, (short)0, acc1[j], false, false);
        }
        __syncthreads();
        if (kt + 1 < KT)
            issue_bfrag_async(Bp, nt0, KT, kt + 1, (kt & 1) ? lds0 : lds1, tid);
    }

    bool keep[16];
#pragma unroll
    for (int mt = 0; mt < 2; ++mt)
#pragma unroll
        for (int v = 0; v < 8; ++v) {
            int m = m0 + mt * 16 + half * 8 + v;
            int s = m & (SEQ - 1);
            keep[mt * 8 + v] = (s < TTEXT) && (s < seq_len[m >> 12]);
        }
#pragma unroll
    for (int j = 0; j < 4; ++j) {
        int n = n0 + j * 16 + col;
#pragma unroll
        for (int v = 0; v < 8; ++v) {
            size_t o0 = (size_t)(m0 + half * 8 + v) * N + n;
            size_t o1 = o0 + (size_t)16 * N;
            X[o0] = keep[v]     ? (X[o0] + acc0[j][v]) : 0.0f;
            X[o1] = keep[8 + v] ? (X[o1] + acc1[j][v]) : 0.0f;
        }
    }
}

// ============================================================
// Stretch-upsample valid prefix to target length (general form)
// ============================================================
__global__ void upsample_kernel(const float* __restrict__ X, const int* __restrict__ seq_len,
                                float* __restrict__ out) {
    size_t i = (size_t)blockIdx.x * 256 + threadIdx.x;     // B*S*D
    int d = (int)(i & (DIM - 1));
    int p = (int)((i >> 9) & (SEQ - 1));
    int b = (int)(i >> 21);
    int sl = seq_len[b];
    int tl = sl < TTEXT ? sl : TTEXT;     // valid text frames (prefix)
    int al = sl;
    int tls = tl > 1 ? tl : 1;
    int base = al / tls;
    int rem = al % tls;
    int split = (tl - rem) * base;
    int bs = base > 1 ? base : 1;
    int j = (p < split) ? (p / bs) : ((tl - rem) + (p - split) / (base + 1));
    if (j < 0) j = 0;
    if (j > SEQ - 1) j = SEQ - 1;
    float v = 0.0f;
    if (p < al && tl > 0) v = X[(((size_t)b * SEQ + j) << 9) + d];
    out[i] = v;
}

// ============================================================
extern "C" void kernel_launch(void* const* d_in, const int* in_sizes, int n_in,
                              void* d_out, int out_size, void* d_ws, size_t ws_size,
                              hipStream_t stream) {
    const int*   text    = (const int*)d_in[0];
    const int*   seq_len = (const int*)d_in[1];
    const float* emb     = (const float*)d_in[2];
    const float* dw_w    = (const float*)d_in[3];
    const float* dw_b    = (const float*)d_in[4];
    const float* ln_g    = (const float*)d_in[5];
    const float* ln_b    = (const float*)d_in[6];
    const float* pw1_w   = (const float*)d_in[7];
    const float* pw1_b   = (const float*)d_in[8];
    const float* grn_g   = (const float*)d_in[9];
    const float* grn_b   = (const float*)d_in[10];
    const float* pw2_w   = (const float*)d_in[11];
    const float* pw2_b   = (const float*)d_in[12];
    float* out = (float*)d_out;

    const size_t ACT  = (size_t)BB * SEQ * DIM;    // 16,777,216 f32
    const size_t ACTI = (size_t)BB * SEQ * INTER;  // 33,554,432 f32

    float*  X   = (float*)d_ws;
    float*  Y   = X + ACT;
    float*  H   = Y + ACT;
    __bf16* W1P = (__bf16*)(H + ACTI);
    __bf16* W2P = W1P + (size_t)LAYERS * DIM * INTER;
    float*  GX  = (float*)(W2P + (size_t)LAYERS * INTER * DIM);
    float*  AL  = GX + BB * INTER;

    for (int l = 0; l < LAYERS; ++l) {
        pack_b_kernel<<<(DIM * INTER + 255) / 256, 256, 0, stream>>>(
            pw1_w + (size_t)l * DIM * INTER, W1P + (size_t)l * DIM * INTER, DIM, INTER);
        pack_b_kernel<<<(INTER * DIM + 255) / 256, 256, 0, stream>>>(
            pw2_w + (size_t)l * INTER * DIM, W2P + (size_t)l * INTER * DIM, INTER, DIM);
    }

    embed_kernel<<<(int)(ACT / 256), 256, 0, stream>>>(text, seq_len, emb, X);

    const int M = BB * SEQ;                       // 32768 rows
    for (int l = 0; l < LAYERS; ++l) {
        dwln_kernel<<<BB * SEQ, 256, 0, stream>>>(
            X, Y, dw_w + (size_t)l * 7 * DIM, dw_b + (size_t)l * DIM,
            ln_g + (size_t)l * DIM, ln_b + (size_t)l * DIM);

        // block = 256M x 64N
        gemm1_gelu_kernel<<<(M / 256) * (INTER / 64), 256, 0, stream>>>(
            Y, W1P + (size_t)l * DIM * INTER, pw1_b + (size_t)l * INTER, H);

        grn_sumsq_kernel<<<BB * (INTER / 64), 256, 0, stream>>>(H, GX);
        grn_alpha_kernel<<<BB, 256, 0, stream>>>(GX, grn_g + (size_t)l * INTER, AL);

        gemm2_res_kernel<<<(M / 256) * (DIM / 64), 256, 0, stream>>>(
            H, W2P + (size_t)l * INTER * DIM, pw2_b + (size_t)l * DIM, X, seq_len,
            AL, grn_b + (size_t)l * INTER);
    }

    upsample_kernel<<<(int)(ACT / 256), 256, 0, stream>>>(X, seq_len, out);
}
